// RecurrentGCN_87926570483780
// MI455X (gfx1250) — compile-verified, matrix operand order checked by reading, not run on hardware
//
#include <hip/hip_runtime.h>
#include <hip/hip_bf16.h>

// ---------------------------------------------------------------------------
// RecurrentGCN (DCRNN x2 + linear + log_softmax) on MI455X / gfx1250.
//
// Math collapses (K=1 DConv, H0=0, log_softmax over size-1 axis):
//   h1 = relu((1 - sigmoid(x@Az1 + bz1)) * tanh(x@Ah1 + bh1))   Az1=Wz1[0][:256]+Wz1[1][:256]
//   h2 = relu((1 - sigmoid(h1@Az2 + bz2)) * tanh(h1@Ah2 + bh2)) Az2=Wz2[0][:32]+Wz2[1][:32]
//   out = log_softmax(h2@Wlin + blin, axis=-1)  == 0 exactly (size-1 axis)
// Wr*/br*, edge_index, edge_weight are dead. Memory-bound on reading x
// (256 MB -> ~11us @ 23.3 TB/s). bf16 WMMA (v_wmma_f32_16x16x32_bf16).
// ---------------------------------------------------------------------------

typedef __attribute__((ext_vector_type(16))) __bf16 v16bf;
typedef __attribute__((ext_vector_type(8)))  float  v8f;

#define NROWS   250000
#define FDIM    256
#define HID     32
#define NTILES  (NROWS / 16)          // 15625, exact
#define WPB     8                     // waves (16-row tiles) per workgroup
#define W1_FRAG_ELEMS (8 * 4 * 32 * 16)  // kk x ntile x lane x 16 = 16384 bf16
#define W2_FRAG_ELEMS (4 * 32 * 16)      //       ntile x lane x 16 =  2048 bf16

// ---------------------------------------------------------------------------
// Prep: combine W[0]+W[1] (live rows only), bf16-convert, and lay out in
// WMMA B-fragment order:  frag[kk][t][lane][i]  with
//   kbase = (lane>=16 ? 8 : 0),  k = kk*32 + kbase + (i&7) + (i&8 ? 16 : 0)
//   col   = (t&1)*16 + (lane&15);  t<2 -> z-weights, t>=2 -> h-weights
// ---------------------------------------------------------------------------
__global__ __launch_bounds__(256) void dcrnn_prep(
    const float* __restrict__ Wz1, const float* __restrict__ Wh1,  // [2,288,32]
    const float* __restrict__ Wz2, const float* __restrict__ Wh2,  // [2, 64,32]
    unsigned short* __restrict__ w1f, unsigned short* __restrict__ w2f)
{
  int e = blockIdx.x * blockDim.x + threadIdx.x;
  if (e < W1_FRAG_ELEMS) {
    int i    = e & 15;
    int lane = (e >> 4) & 31;
    int t    = (e >> 9) & 3;
    int kk   = e >> 11;                       // 0..7
    int k = kk * 32 + ((lane & 16) ? 8 : 0) + (i & 7) + ((i & 8) ? 16 : 0);
    int c = ((t & 1) ? 16 : 0) + (lane & 15);
    const float* W = (t < 2) ? Wz1 : Wh1;     // rows k < 256 only (H0 rows dead)
    float v = W[k * HID + c] + W[288 * HID + k * HID + c];
    __bf16 bv = (__bf16)v;
    w1f[e] = __builtin_bit_cast(unsigned short, bv);
    return;
  }
  int e2 = e - W1_FRAG_ELEMS;
  if (e2 < W2_FRAG_ELEMS) {
    int i    = e2 & 15;
    int lane = (e2 >> 4) & 31;
    int t    = (e2 >> 9) & 3;
    int k = ((lane & 16) ? 8 : 0) + (i & 7) + ((i & 8) ? 16 : 0);  // K=32
    int c = ((t & 1) ? 16 : 0) + (lane & 15);
    const float* W = (t < 2) ? Wz2 : Wh2;     // rows k < 32 only
    float v = W[k * HID + c] + W[64 * HID + k * HID + c];
    __bf16 bv = (__bf16)v;
    w2f[e2] = __builtin_bit_cast(unsigned short, bv);
  }
}

static __device__ __forceinline__ float sigmf(float v) {
  return 1.0f / (1.0f + expf(-v));
}

// ---------------------------------------------------------------------------
// Fused pipeline: one wave per 16-row tile of x.
// ---------------------------------------------------------------------------
__global__ __launch_bounds__(256) void dcrnn_fused(
    const float* __restrict__ x,
    const float* __restrict__ bz1, const float* __restrict__ bh1,
    const float* __restrict__ bz2, const float* __restrict__ bh2,
    const float* __restrict__ wlin, const float* __restrict__ blin,
    const unsigned short* __restrict__ w1f,
    const unsigned short* __restrict__ w2f,
    float* __restrict__ out)
{
  __shared__ __align__(32) __bf16 sW1[W1_FRAG_ELEMS];   // 32 KB
  __shared__ __align__(32) __bf16 sW2[W2_FRAG_ELEMS];   //  4 KB
  __shared__ __align__(32) __bf16 sH[WPB][16 * HID];    //  8 KB

  const int tid = threadIdx.x;

  // Stage weight fragments into LDS (b128 copies).
  {
    const uint4* s1 = (const uint4*)w1f;
    uint4*       d1 = (uint4*)sW1;
    #pragma unroll 2
    for (int i = tid; i < W1_FRAG_ELEMS / 8; i += 256) d1[i] = s1[i];
    const uint4* s2 = (const uint4*)w2f;
    uint4*       d2 = (uint4*)sW2;
    if (tid < W2_FRAG_ELEMS / 8) d2[tid] = s2[tid];
  }
  __syncthreads();

  const int wave = tid >> 5;
  const int lane = tid & 31;
  const int tile = blockIdx.x * WPB + wave;
  if (tile >= NTILES) return;                 // wave-uniform: EXEC stays full

  const int nsub  = lane & 15;                // A: row M / B: col N / C: col N
  const int half  = lane >> 4;                // 0: K 0..7/16..23, 1: K 8..15/24..31
  const int kbase = half ? 8 : 0;

  const float* xrow = x + (size_t)(tile * 16 + nsub) * FDIM;

  // ---- Layer 1: [16x256] @ [256x64] (cols 0..31=z, 32..63=h) ----
  v8f az0 = {}, az1 = {}, ah0 = {}, ah1 = {};
  #pragma unroll 2
  for (int kk = 0; kk < 8; ++kk) {
    const int k0 = kk * 32 + kbase;
    v16bf a;
    #pragma unroll
    for (int i = 0; i < 8; ++i) {
      a[i]     = (__bf16)xrow[k0 + i];
      a[i + 8] = (__bf16)xrow[k0 + 16 + i];
    }
    const v16bf* bf = (const v16bf*)&sW1[(size_t)kk * 4 * 32 * 16];
    v16bf b0 = bf[0 * 32 + lane];
    v16bf b1 = bf[1 * 32 + lane];
    v16bf b2 = bf[2 * 32 + lane];
    v16bf b3 = bf[3 * 32 + lane];
    az0 = __builtin_amdgcn_wmma_f32_16x16x32_bf16(false, a, false, b0, (short)0, az0, false, false);
    az1 = __builtin_amdgcn_wmma_f32_16x16x32_bf16(false, a, false, b1, (short)0, az1, false, false);
    ah0 = __builtin_amdgcn_wmma_f32_16x16x32_bf16(false, a, false, b2, (short)0, ah0, false, false);
    ah1 = __builtin_amdgcn_wmma_f32_16x16x32_bf16(false, a, false, b3, (short)0, ah1, false, false);
  }

  // Epilogue 1: h1 = relu((1-sigmoid(z+bz))*tanh(h+bh)); write to LDS row-major.
  {
    const float bzv0 = bz1[nsub],      bzv1 = bz1[16 + nsub];
    const float bhv0 = bh1[nsub],      bhv1 = bh1[16 + nsub];
    #pragma unroll
    for (int r = 0; r < 8; ++r) {
      const int row = r + (half ? 8 : 0);     // C layout: lanes 16-31 hold M+8
      float z0 = sigmf(az0[r] + bzv0);
      float z1 = sigmf(az1[r] + bzv1);
      float t0 = tanhf(ah0[r] + bhv0);
      float t1 = tanhf(ah1[r] + bhv1);
      float o0 = fmaxf((1.0f - z0) * t0, 0.0f);
      float o1 = fmaxf((1.0f - z1) * t1, 0.0f);
      sH[wave][row * HID + nsub]      = (__bf16)o0;
      sH[wave][row * HID + 16 + nsub] = (__bf16)o1;
    }
  }
  // Same-wave LDS write -> cross-lane read: drain DS counter explicitly.
  asm volatile("s_wait_dscnt 0" ::: "memory");

  // ---- Layer 2: [16x32] @ [32x64], single K-step ----
  v16bf a2;
  #pragma unroll
  for (int i = 0; i < 8; ++i) {
    a2[i]     = sH[wave][nsub * HID + kbase + i];
    a2[i + 8] = sH[wave][nsub * HID + kbase + 16 + i];
  }
  const v16bf* b2f = (const v16bf*)sW2;
  v16bf c0 = b2f[0 * 32 + lane];
  v16bf c1 = b2f[1 * 32 + lane];
  v16bf c2 = b2f[2 * 32 + lane];
  v16bf c3 = b2f[3 * 32 + lane];
  v8f zz0 = {}, zz1 = {}, hh0 = {}, hh1 = {};
  zz0 = __builtin_amdgcn_wmma_f32_16x16x32_bf16(false, a2, false, c0, (short)0, zz0, false, false);
  zz1 = __builtin_amdgcn_wmma_f32_16x16x32_bf16(false, a2, false, c1, (short)0, zz1, false, false);
  hh0 = __builtin_amdgcn_wmma_f32_16x16x32_bf16(false, a2, false, c2, (short)0, hh0, false, false);
  hh1 = __builtin_amdgcn_wmma_f32_16x16x32_bf16(false, a2, false, c3, (short)0, hh1, false, false);

  // Epilogue 2 + row reduction:  v = sum_c h2[r][c]*Wlin[c] + blin.
  const float bzv0 = bz2[nsub], bzv1 = bz2[16 + nsub];
  const float bhv0 = bh2[nsub], bhv1 = bh2[16 + nsub];
  const float wl0  = wlin[nsub], wl1 = wlin[16 + nsub];
  const float bl   = blin[0];
  float s[8];
  #pragma unroll
  for (int r = 0; r < 8; ++r) {
    float z0 = sigmf(zz0[r] + bzv0);
    float z1 = sigmf(zz1[r] + bzv1);
    float t0 = tanhf(hh0[r] + bhv0);
    float t1 = tanhf(hh1[r] + bhv1);
    float o0 = fmaxf((1.0f - z0) * t0, 0.0f);
    float o1 = fmaxf((1.0f - z1) * t1, 0.0f);
    s[r] = o0 * wl0 + o1 * wl1;
  }
  // XOR-reduce across the 16 lanes of each half (masks < 16 keep halves apart).
  #pragma unroll
  for (int m = 1; m < 16; m <<= 1) {
    #pragma unroll
    for (int r = 0; r < 8; ++r) s[r] += __shfl_xor(s[r], m, 32);
  }

  // log_softmax over a size-1 axis: x - logsumexp(x) = v - v = 0 for finite v.
  // (v - v keeps the full WMMA pipeline live; bit-exact with the reference.)
  if (nsub == 0) {                            // lanes 0 (rows 0-7) and 16 (rows 8-15)
    const int rowbase = tile * 16 + (half ? 8 : 0);
    #pragma unroll
    for (int r = 0; r < 8; ++r) {
      float v = s[r] + bl;
      out[rowbase + r] = v - v;
    }
  }
}

extern "C" void kernel_launch(void* const* d_in, const int* in_sizes, int n_in,
                              void* d_out, int out_size, void* d_ws, size_t ws_size,
                              hipStream_t stream) {
  (void)in_sizes; (void)n_in; (void)out_size; (void)ws_size;
  const float* x    = (const float*)d_in[0];
  // d_in[1] edge_index, d_in[2] edge_weight: dead code when K==1.
  const float* Wz1  = (const float*)d_in[3];
  const float* bz1  = (const float*)d_in[4];
  // d_in[5]/[6] Wr1/br1: reset gate multiplies H0==0 -> dead.
  const float* Wh1  = (const float*)d_in[7];
  const float* bh1  = (const float*)d_in[8];
  const float* Wz2  = (const float*)d_in[9];
  const float* bz2  = (const float*)d_in[10];
  // d_in[11]/[12] Wr2/br2: dead.
  const float* Wh2  = (const float*)d_in[13];
  const float* bh2  = (const float*)d_in[14];
  const float* wlin = (const float*)d_in[15];
  const float* blin = (const float*)d_in[16];

  unsigned short* w1f = (unsigned short*)d_ws;                 // 16384 bf16
  unsigned short* w2f = w1f + W1_FRAG_ELEMS;                   //  2048 bf16

  const int prep_elems = W1_FRAG_ELEMS + W2_FRAG_ELEMS;        // 18432
  dcrnn_prep<<<(prep_elems + 255) / 256, 256, 0, stream>>>(
      Wz1, Wh1, Wz2, Wh2, w1f, w2f);

  const int nblocks = (NTILES + WPB - 1) / WPB;                // 1954
  dcrnn_fused<<<nblocks, 256, 0, stream>>>(
      x, bz1, bh1, bz2, bh2, wlin, blin, w1f, w2f, (float*)d_out);
}
